// GPT2Attention_27822798143707
// MI455X (gfx1250) — matrix operand order repaired and checked by reading, not compiled, not run on hardware
//
#include <hip/hip_runtime.h>
#include <hip/hip_bf16.h>

typedef __attribute__((ext_vector_type(16))) __bf16 v16bf;
typedef __attribute__((ext_vector_type(8)))  __bf16 v8bf;
typedef __attribute__((ext_vector_type(8)))  float  v8f;

#define BB 8
#define TT 1024
#define DD 768
#define NH 12
#define HD 64

__device__ __forceinline__ v8f wmma_bf16(v16bf a, v16bf b, v8f c) {
    return __builtin_amdgcn_wmma_f32_16x16x32_bf16(false, a, false, b, (short)0, c,
                                                   false, false);
}

union U16 { v16bf v; v8bf h[2]; };

// ---------------------------------------------------------------------------
// fp32 -> bf16 elementwise convert
// ---------------------------------------------------------------------------
__global__ void conv_bf16(const float* __restrict__ s, __bf16* __restrict__ d, int n) {
    int i = blockIdx.x * blockDim.x + threadIdx.x;
    if (i < n) d[i] = (__bf16)s[i];
}

// fp32 [K,N] -> bf16 transposed [N,K]  (i indexes output: i = n*K + k)
__global__ void conv_t_bf16(const float* __restrict__ W, __bf16* __restrict__ Wt,
                            int K, int N, int total) {
    int i = blockIdx.x * blockDim.x + threadIdx.x;
    if (i >= total) return;
    int n = i / K, k = i - n * K;
    Wt[i] = (__bf16)W[k * N + n];
}

// ---------------------------------------------------------------------------
// One wave computes a 16x64 C tile: one A fragment reused across 4 B columns.
// Explicitly double-buffered: fragments for step kk+32 are fetched while the
// four WMMAs for step kk execute, so v_wmma never waits on a just-issued load.
// A[M,K] bf16 row-major, Bt[N,K] bf16 (transposed weights). K must be a
// multiple of 32 and >= 64; call sites pass the compile-time constant DD.
// ---------------------------------------------------------------------------
__device__ __forceinline__ void wmma_tile4(const __bf16* __restrict__ A,
                                           const __bf16* __restrict__ Bt,
                                           int K, int m0, int n0, int lane,
                                           v8f acc[4]) {
    int lrow = lane & 15;
    int lhi  = lane >> 4;
    int koff = lhi * 16;    // A fragment: 16 contiguous halves at koff
    int kb   = lhi * 16;    // B fragment: 16 contiguous halves at kb
    const __bf16* arow = A + (size_t)(m0 + lrow) * K + koff;
    const __bf16* brow = Bt + (size_t)(n0 + lrow) * K + kb;

    U16 a_cur, a_nxt;
    v16bf b_cur[4], b_nxt[4];

    a_cur.h[0] = *(const v8bf*)(arow);
    a_cur.h[1] = *(const v8bf*)(arow + 8);
#pragma unroll
    for (int g = 0; g < 4; ++g)
        b_cur[g] = *(const v16bf*)(brow + (size_t)g * 16 * K);

#pragma unroll
    for (int kk = 0; kk < K; kk += 32) {
        int kn = kk + 32;
        if (kn < K) {   // folds away: K is a propagated constant at call sites
            a_nxt.h[0] = *(const v8bf*)(arow + kn);
            a_nxt.h[1] = *(const v8bf*)(arow + kn + 8);
#pragma unroll
            for (int g = 0; g < 4; ++g)
                b_nxt[g] = *(const v16bf*)(brow + (size_t)g * 16 * K + kn);
        }
#pragma unroll
        for (int g = 0; g < 4; ++g)
            acc[g] = wmma_bf16(a_cur.v, b_cur[g], acc[g]);
        a_cur = a_nxt;
#pragma unroll
        for (int g = 0; g < 4; ++g) b_cur[g] = b_nxt[g];
    }
}

// ---------------------------------------------------------------------------
// QKV projection: [8192,768] x [768,2304] + b_attn, 16x64 tile per wave.
// Q,K scattered to [B,H,T,64] bf16; V scattered transposed to [B,H,64,T].
// ---------------------------------------------------------------------------
__global__ __launch_bounds__(128) void qkv_gemm(const __bf16* __restrict__ Xbf,
                                                const __bf16* __restrict__ Wt,
                                                const float* __restrict__ b_attn,
                                                __bf16* __restrict__ Qbf,
                                                __bf16* __restrict__ Kbf,
                                                __bf16* __restrict__ Vt) {
    const int NT = (3 * DD) / 64;  // 36 tile cols
    int wid  = (blockIdx.x * blockDim.x + threadIdx.x) >> 5;
    int lane = threadIdx.x & 31;
    int mt = wid / NT, nt = wid - mt * NT;
    int m0 = mt * 16, n0 = nt * 64;

    v8f acc[4] = {v8f{}, v8f{}, v8f{}, v8f{}};
    wmma_tile4(Xbf, Wt, DD, m0, n0, lane, acc);

    int lrow  = lane & 15;
    int rbase = m0 + ((lane >> 4) << 3);
#pragma unroll
    for (int g = 0; g < 4; ++g) {
        int col  = n0 + g * 16 + lrow;
        float bias = b_attn[col];
        int part   = col / DD;
        int within = col - part * DD;
        int head   = within >> 6;
        int dim    = within & 63;
#pragma unroll
        for (int v = 0; v < 8; ++v) {
            int row = rbase + v;
            int b = row >> 10, t = row & (TT - 1);
            __bf16 val = (__bf16)(acc[g][v] + bias);
            if (part == 0)
                Qbf[(((b * NH + head) << 10) + t) * HD + dim] = val;
            else if (part == 1)
                Kbf[(((b * NH + head) << 10) + t) * HD + dim] = val;
            else
                Vt[((b * NH + head) * HD + dim) * TT + t] = val;
        }
    }
}

// ---------------------------------------------------------------------------
// Flash attention: one wave per (b,h, 32-query strip-pair); 32 keys per block.
// Both 16-row strips share the loaded K and V fragments.
// ---------------------------------------------------------------------------
__global__ __launch_bounds__(128) void attn_kernel(const __bf16* __restrict__ Qbf,
                                                   const __bf16* __restrict__ Kbf,
                                                   const __bf16* __restrict__ Vt,
                                                   __bf16* __restrict__ Obf) {
    __shared__ __align__(32) __bf16 lds[4][2][16 * 32];
    int lane = threadIdx.x & 31;
    int winb = threadIdx.x >> 5;
    int wid  = blockIdx.x * 4 + winb;
    int qp = wid & 31;     // 32 query strip-pairs per head (32 rows each)
    int bh = wid >> 5;     // 0..95

    const __bf16* Qp = Qbf + (size_t)bh * TT * HD;
    const __bf16* Kp = Kbf + (size_t)bh * TT * HD;
    const __bf16* Vp = Vt  + (size_t)bh * HD * TT;

    int q0   = qp * 32;
    int lrow = lane & 15;
    int lhi  = lane >> 4;
    int koff = lhi * 16;

    // Q A-fragments: 2 strips x 2 K=32 chunks
    U16 qf[2][2];
#pragma unroll
    for (int s = 0; s < 2; ++s)
#pragma unroll
        for (int c = 0; c < 2; ++c) {
            const __bf16* base = Qp + (q0 + s * 16 + lrow) * HD + c * 32 + koff;
            qf[s][c].h[0] = *(const v8bf*)(base);
            qf[s][c].h[1] = *(const v8bf*)(base + 8);
        }

    v8f O[2][4] = {};
    float m[2][8], l[2][8];
#pragma unroll
    for (int s = 0; s < 2; ++s)
#pragma unroll
        for (int v = 0; v < 8; ++v) { m[s][v] = -1e30f; l[s][v] = 0.0f; }

    int rbl = lhi * 8;     // local row within strip for accumulator slot v
    int nkb = qp + 1;      // causal: key blocks covering 0..q0+31

    for (int kb = 0; kb < nkb; ++kb) {
        int k0 = kb * 32;
        // K fragments, shared by both strips: [chunk][key-half]
        v16bf bk[2][2];
#pragma unroll
        for (int c = 0; c < 2; ++c)
#pragma unroll
            for (int hk = 0; hk < 2; ++hk)
                bk[c][hk] = *(const v16bf*)(Kp + (k0 + hk * 16 + lrow) * HD +
                                            c * 32 + lhi * 16);

        U16 pf[2];
#pragma unroll
        for (int s = 0; s < 2; ++s) {
            v8f S0 = {}, S1 = {};
#pragma unroll
            for (int c = 0; c < 2; ++c) {
                S0 = wmma_bf16(qf[s][c].v, bk[c][0], S0);
                S1 = wmma_bf16(qf[s][c].v, bk[c][1], S1);
            }

            const float scale = 0.125f;    // 1/sqrt(64)
            int key0 = k0 + lrow, key1 = k0 + 16 + lrow;
            int rowb = q0 + s * 16 + rbl;
            float mn[8];
#pragma unroll
            for (int v = 0; v < 8; ++v) {
                int row  = rowb + v;
                float s0 = (key0 <= row) ? S0[v] * scale : -1e30f;
                float s1 = (key1 <= row) ? S1[v] * scale : -1e30f;
                S0[v] = s0; S1[v] = s1;
                mn[v] = fmaxf(s0, s1);
            }
#pragma unroll
            for (int off = 1; off < 16; off <<= 1)
#pragma unroll
                for (int v = 0; v < 8; ++v)
                    mn[v] = fmaxf(mn[v], __shfl_xor(mn[v], off, 32));

            float p0[8], p1[8], alpha[8], rs[8];
#pragma unroll
            for (int v = 0; v < 8; ++v) {
                float mnew = fmaxf(m[s][v], mn[v]);
                alpha[v] = __expf(m[s][v] - mnew);
                m[s][v] = mnew;
                p0[v] = __expf(S0[v] - mnew);
                p1[v] = __expf(S1[v] - mnew);
                rs[v] = p0[v] + p1[v];
            }
#pragma unroll
            for (int off = 1; off < 16; off <<= 1)
#pragma unroll
                for (int v = 0; v < 8; ++v)
                    rs[v] += __shfl_xor(rs[v], off, 32);
#pragma unroll
            for (int v = 0; v < 8; ++v) {
                l[s][v] = l[s][v] * alpha[v] + rs[v];
#pragma unroll
                for (int f = 0; f < 4; ++f) O[s][f][v] *= alpha[v];
            }

            // stage P (16q x 32k) through LDS: C-layout -> A-fragment layout
            __bf16* lp = lds[winb][s];
#pragma unroll
            for (int v = 0; v < 8; ++v) {
                int r = lhi * 8 + v;
                lp[r * 32 + lrow]      = (__bf16)p0[v];
                lp[r * 32 + 16 + lrow] = (__bf16)p1[v];
            }
            asm volatile("s_wait_dscnt 0" ::: "memory");
            const __bf16* pbase = lp + lrow * 32 + koff;
            pf[s].h[0] = *(const v8bf*)(pbase);
            pf[s].h[1] = *(const v8bf*)(pbase + 8);
        }

        // P(16x32) @ V(32x64): V fragments shared by both strips
#pragma unroll
        for (int f = 0; f < 4; ++f) {
            v16bf vf = *(const v16bf*)(Vp + (f * 16 + lrow) * TT + k0 + lhi * 16);
            O[0][f] = wmma_bf16(pf[0].v, vf, O[0][f]);
            O[1][f] = wmma_bf16(pf[1].v, vf, O[1][f]);
        }
    }

    int bIdx = bh / NH, h = bh - bIdx * NH;
#pragma unroll
    for (int s = 0; s < 2; ++s)
#pragma unroll
        for (int v = 0; v < 8; ++v) {
            float inv = 1.0f / l[s][v];
            int t = q0 + s * 16 + rbl + v;
            __bf16* orow = Obf + ((size_t)((bIdx << 10) + t)) * DD + h * HD;
            orow[ 0 + lrow] = (__bf16)(O[s][0][v] * inv);
            orow[16 + lrow] = (__bf16)(O[s][1][v] * inv);
            orow[32 + lrow] = (__bf16)(O[s][2][v] * inv);
            orow[48 + lrow] = (__bf16)(O[s][3][v] * inv);
        }
}

// ---------------------------------------------------------------------------
// Output projection: [8192,768] x [768,768] + b_proj -> fp32 d_out
// ---------------------------------------------------------------------------
__global__ __launch_bounds__(128) void proj_gemm(const __bf16* __restrict__ Obf,
                                                 const __bf16* __restrict__ Wt,
                                                 const float* __restrict__ b_proj,
                                                 float* __restrict__ out) {
    const int NT = DD / 64;  // 12 tile cols
    int wid  = (blockIdx.x * blockDim.x + threadIdx.x) >> 5;
    int lane = threadIdx.x & 31;
    int mt = wid / NT, nt = wid - mt * NT;
    int m0 = mt * 16, n0 = nt * 64;

    v8f acc[4] = {v8f{}, v8f{}, v8f{}, v8f{}};
    wmma_tile4(Obf, Wt, DD, m0, n0, lane, acc);

    int lrow  = lane & 15;
    int rbase = m0 + ((lane >> 4) << 3);
#pragma unroll
    for (int g = 0; g < 4; ++g) {
        int col = n0 + g * 16 + lrow;
        float bias = b_proj[col];
#pragma unroll
        for (int v = 0; v < 8; ++v)
            out[(size_t)(rbase + v) * DD + col] = acc[g][v] + bias;
    }
}

// ---------------------------------------------------------------------------
extern "C" void kernel_launch(void* const* d_in, const int* in_sizes, int n_in,
                              void* d_out, int out_size, void* d_ws, size_t ws_size,
                              hipStream_t stream) {
    const float* x      = (const float*)d_in[0];  // [8,1024,768]
    const float* W_attn = (const float*)d_in[1];  // [768,2304]
    const float* b_attn = (const float*)d_in[2];  // [2304]
    const float* W_proj = (const float*)d_in[3];  // [768,768]
    const float* b_proj = (const float*)d_in[4];  // [768]
    float* out = (float*)d_out;

    const size_t nX   = (size_t)BB * TT * DD;          // 6,291,456
    const size_t nWA  = (size_t)DD * 3 * DD;           // 1,769,472
    const size_t nWP  = (size_t)DD * DD;               //   589,824
    const size_t nQKV = (size_t)BB * NH * TT * HD;     // 6,291,456 per tensor

    char* ws = (char*)d_ws;
    size_t off = 0;
    auto take = [&](size_t elems) { void* p = ws + off; off += ((elems * 2 + 255) & ~(size_t)255); return p; };
    __bf16* Xbf  = (__bf16*)take(nX);
    __bf16* WAt  = (__bf16*)take(nWA);   // [2304,768] transposed
    __bf16* WPt  = (__bf16*)take(nWP);   // [768,768]  transposed
    __bf16* Qbf  = (__bf16*)take(nQKV);  // [B,H,T,64]
    __bf16* Kbf  = (__bf16*)take(nQKV);  // [B,H,T,64]
    __bf16* Vt   = (__bf16*)take(nQKV);  // [B,H,64,T]
    __bf16* Obf  = (__bf16*)take(nX);    // [8192,768]

    // 1. conversions
    conv_bf16<<<(nX + 255) / 256, 256, 0, stream>>>(x, Xbf, (int)nX);
    conv_t_bf16<<<(nWA + 255) / 256, 256, 0, stream>>>(W_attn, WAt, DD, 3 * DD, (int)nWA);
    conv_t_bf16<<<(nWP + 255) / 256, 256, 0, stream>>>(W_proj, WPt, DD, DD, (int)nWP);

    // 2. QKV GEMM: 512 x 36 (16x64) tiles, 4 waves/block
    qkv_gemm<<<(512 * 36) / 4, 128, 0, stream>>>(Xbf, WAt, b_attn, Qbf, Kbf, Vt);

    // 3. flash attention: 96 bh * 32 strip-pairs = 3072 waves
    attn_kernel<<<3072 / 4, 128, 0, stream>>>(Qbf, Kbf, Vt, Obf);

    // 4. output projection: 512 x 12 (16x64) tiles
    proj_gemm<<<(512 * 12) / 4, 128, 0, stream>>>(Obf, WPt, b_proj, out);
}
